// ResidualVQ_51238959841480
// MI455X (gfx1250) — compile-verified, matrix operand order checked by reading, not run on hardware
//
#include <hip/hip_runtime.h>

// ---------------------------------------------------------------------------
// Residual VQ for MI455X (gfx1250, wave32, WMMA).
// Distances via bf16 hi/lo split WMMA (v_wmma_f32_16x16x32_bf16), everything
// else (gather, residual update, loss) exact fp32.
// ---------------------------------------------------------------------------

typedef __attribute__((ext_vector_type(16))) __bf16 v16bf;
typedef __attribute__((ext_vector_type(8)))  float  v8f;

#define B_   8
#define N_   1024
#define D_   512
#define C_   1024
#define NQ   8
#define ROWS (B_ * N_)        // 8192
#define DU   (D_ / 2)         // 256 uints (bf16 pairs) per row

union Frag { v16bf v; unsigned int u[8]; };

// Load a 16-bf16-per-lane A/B fragment following the CDNA5 16-bit 16x32 layout:
// lanes 0-15 hold K = 0..7 and 16..23; lanes 16-31 hold K = 8..15 and 24..31.
// row points at a 512-element bf16 row (as packed uint pairs), ktBase = kt*16
// (uint units), kb2 = 0 or 4 depending on lane half.
__device__ __forceinline__ v16bf load_frag(const unsigned int* __restrict__ row,
                                           int ktBase, int kb2) {
  Frag f;
#pragma unroll
  for (int j = 0; j < 4; ++j) f.u[j] = row[ktBase + kb2 + j];
#pragma unroll
  for (int j = 0; j < 4; ++j) f.u[4 + j] = row[ktBase + 8 + kb2 + j];
  return f.v;
}

__device__ __forceinline__ float trunc_bf(float a) {
  return __uint_as_float(__float_as_uint(a) & 0xFFFF0000u);
}
__device__ __forceinline__ unsigned int pack_bf(float a, float b) {
  return (__float_as_uint(a) >> 16) | ((__float_as_uint(b) >> 16) << 16);
}

// ---------------------------------------------------------------------------
// init: residual = x, qout = 0, build bf16 hi/lo mirrors of residual.
// grid = ROWS blocks x 128 threads (4 floats each).
__global__ void __launch_bounds__(128)
rvq_init(const float* __restrict__ x, float* __restrict__ residual,
         unsigned int* __restrict__ res_hi, unsigned int* __restrict__ res_lo,
         float* __restrict__ qout) {
  const int row = blockIdx.x;
  const int t   = threadIdx.x;
  const float4* xr = (const float4*)(x + (size_t)row * D_);
  float4* rr = (float4*)(residual + (size_t)row * D_);
  float4* qr = (float4*)(qout     + (size_t)row * D_);
  unsigned int* hr = res_hi + (size_t)row * DU;
  unsigned int* lr = res_lo + (size_t)row * DU;

  float4 v = xr[t];
  rr[t] = v;
  qr[t] = make_float4(0.f, 0.f, 0.f, 0.f);
  float l0 = v.x - trunc_bf(v.x), l1 = v.y - trunc_bf(v.y);
  float l2 = v.z - trunc_bf(v.z), l3 = v.w - trunc_bf(v.w);
  hr[2 * t]     = pack_bf(v.x, v.y);
  hr[2 * t + 1] = pack_bf(v.z, v.w);
  lr[2 * t]     = pack_bf(l0, l1);
  lr[2 * t + 1] = pack_bf(l2, l3);
}

// ---------------------------------------------------------------------------
// prep: bf16 hi/lo mirrors of all codebooks + per-code squared norms.
// grid = NQ*C_ blocks x 128 threads.
__global__ void __launch_bounds__(128)
rvq_prep(const float* __restrict__ cb, unsigned int* __restrict__ cb_hi,
         unsigned int* __restrict__ cb_lo, float* __restrict__ e2) {
  const int row = blockIdx.x;        // q*C_ + c
  const int t   = threadIdx.x;
  const float4* cr = (const float4*)(cb + (size_t)row * D_);
  unsigned int* hr = cb_hi + (size_t)row * DU;
  unsigned int* lr = cb_lo + (size_t)row * DU;

  float4 v = cr[t];
  float l0 = v.x - trunc_bf(v.x), l1 = v.y - trunc_bf(v.y);
  float l2 = v.z - trunc_bf(v.z), l3 = v.w - trunc_bf(v.w);
  hr[2 * t]     = pack_bf(v.x, v.y);
  hr[2 * t + 1] = pack_bf(v.z, v.w);
  lr[2 * t]     = pack_bf(l0, l1);
  lr[2 * t + 1] = pack_bf(l2, l3);

  __shared__ float sm[128];
  sm[t] = v.x * v.x + v.y * v.y + v.z * v.z + v.w * v.w;
  __syncthreads();
#pragma unroll
  for (int s = 64; s > 0; s >>= 1) {
    if (t < s) sm[t] += sm[t + s];
    __syncthreads();
  }
  if (t == 0) e2[row] = sm[0];
}

// ---------------------------------------------------------------------------
// argmin: one block = 32 residual rows (2 M-tiles), 8 waves split C into
// 8 c-tiles each. dots via bf16-split WMMA; argmin of (||e||^2 - 2*dot).
// grid = ROWS/32 blocks x 256 threads.
__global__ void __launch_bounds__(256)
rvq_argmin(const unsigned int* __restrict__ res_hi,
           const unsigned int* __restrict__ res_lo,
           const unsigned int* __restrict__ cbq_hi,
           const unsigned int* __restrict__ cbq_lo,
           const float* __restrict__ e2q,
           int* __restrict__ idx_ws, float* __restrict__ idx_out, int q) {
  const int wave = threadIdx.x >> 5;
  const int lane = threadIdx.x & 31;
  const int col  = lane & 15;
  const int kb2  = (lane & 16) ? 4 : 0;
  const int row0 = blockIdx.x * 32 + col;   // M-tile 0 row for this lane
  const int row1 = row0 + 16;               // M-tile 1 row

  const unsigned int* aH0 = res_hi + (size_t)row0 * DU;
  const unsigned int* aL0 = res_lo + (size_t)row0 * DU;
  const unsigned int* aH1 = res_hi + (size_t)row1 * DU;
  const unsigned int* aL1 = res_lo + (size_t)row1 * DU;

  const unsigned int* bH[8];
  const unsigned int* bL[8];
  int cidx[8];
#pragma unroll
  for (int t = 0; t < 8; ++t) {
    int c = (wave + t * 8) * 16 + col;      // ascending in t -> first-min wins
    cidx[t] = c;
    bH[t] = cbq_hi + (size_t)c * DU;
    bL[t] = cbq_lo + (size_t)c * DU;
  }

  v8f acc0[8], acc1[8];
  v8f z = {0.f, 0.f, 0.f, 0.f, 0.f, 0.f, 0.f, 0.f};
#pragma unroll
  for (int t = 0; t < 8; ++t) { acc0[t] = z; acc1[t] = z; }

  for (int kt = 0; kt < 16; ++kt) {         // K = 512 in chunks of 32
    const int kb = kt * 16;
    v16bf a0h = load_frag(aH0, kb, kb2);
    v16bf a0l = load_frag(aL0, kb, kb2);
    v16bf a1h = load_frag(aH1, kb, kb2);
    v16bf a1l = load_frag(aL1, kb, kb2);
#pragma unroll
    for (int t = 0; t < 8; ++t) {
      v16bf bh = load_frag(bH[t], kb, kb2);
      v16bf bl = load_frag(bL[t], kb, kb2);
      // dot(r, e) ~= hi*hi + hi*lo + lo*hi  (lo*lo ~ 2^-32, dropped)
      acc0[t] = __builtin_amdgcn_wmma_f32_16x16x32_bf16(false, a0h, false, bh, (short)0, acc0[t], false, false);
      acc0[t] = __builtin_amdgcn_wmma_f32_16x16x32_bf16(false, a0h, false, bl, (short)0, acc0[t], false, false);
      acc0[t] = __builtin_amdgcn_wmma_f32_16x16x32_bf16(false, a0l, false, bh, (short)0, acc0[t], false, false);
      acc1[t] = __builtin_amdgcn_wmma_f32_16x16x32_bf16(false, a1h, false, bh, (short)0, acc1[t], false, false);
      acc1[t] = __builtin_amdgcn_wmma_f32_16x16x32_bf16(false, a1h, false, bl, (short)0, acc1[t], false, false);
      acc1[t] = __builtin_amdgcn_wmma_f32_16x16x32_bf16(false, a1l, false, bh, (short)0, acc1[t], false, false);
    }
  }

  // Per-lane running argmin over this lane's c-column family.
  // C/D layout: element i of acc -> row m = i + 8*(lane>=16), column = col.
  float minv0[8], minv1[8];
  int   mini0[8], mini1[8];
#pragma unroll
  for (int i = 0; i < 8; ++i) {
    minv0[i] = 3.4e38f; minv1[i] = 3.4e38f; mini0[i] = 0; mini1[i] = 0;
  }
#pragma unroll
  for (int t = 0; t < 8; ++t) {
    float ev = e2q[cidx[t]];
#pragma unroll
    for (int i = 0; i < 8; ++i) {
      float d0 = ev - 2.f * acc0[t][i];
      if (d0 < minv0[i]) { minv0[i] = d0; mini0[i] = cidx[t]; }
      float d1 = ev - 2.f * acc1[t][i];
      if (d1 < minv1[i]) { minv1[i] = d1; mini1[i] = cidx[t]; }
    }
  }

  // Cross-lane reduce inside each 16-lane half (row sets are disjoint per half).
#pragma unroll
  for (int m = 1; m <= 8; m <<= 1) {
#pragma unroll
    for (int i = 0; i < 8; ++i) {
      float ov = __shfl_xor(minv0[i], m, 32);
      int   oi = __shfl_xor(mini0[i], m, 32);
      if (ov < minv0[i] || (ov == minv0[i] && oi < mini0[i])) { minv0[i] = ov; mini0[i] = oi; }
      ov = __shfl_xor(minv1[i], m, 32);
      oi = __shfl_xor(mini1[i], m, 32);
      if (ov < minv1[i] || (ov == minv1[i] && oi < mini1[i])) { minv1[i] = ov; mini1[i] = oi; }
    }
  }

  __shared__ float s_val[8][32];
  __shared__ int   s_idx[8][32];
  if (col == 0) {                            // lanes 0 and 16 hold results
    const int mb = (lane & 16) ? 8 : 0;
#pragma unroll
    for (int i = 0; i < 8; ++i) {
      s_val[wave][mb + i]      = minv0[i];
      s_idx[wave][mb + i]      = mini0[i];
      s_val[wave][16 + mb + i] = minv1[i];
      s_idx[wave][16 + mb + i] = mini1[i];
    }
  }
  __syncthreads();
  if (threadIdx.x < 32) {                    // final inter-wave reduce (fixed order)
    const int m = threadIdx.x;
    float bv = s_val[0][m];
    int   bi = s_idx[0][m];
#pragma unroll
    for (int w = 1; w < 8; ++w) {
      float v = s_val[w][m];
      int   i2 = s_idx[w][m];
      if (v < bv || (v == bv && i2 < bi)) { bv = v; bi = i2; }
    }
    const int row = blockIdx.x * 32 + m;
    idx_ws[row * NQ + q]  = bi;
    idx_out[row * NQ + q] = (float)bi;       // output dtype is float
  }
}

// ---------------------------------------------------------------------------
// update: exact fp32 gather, residual -= e, qout += e, refresh bf16 mirrors,
// per-row loss partial = sum (r - e)^2.  grid = ROWS blocks x 128 threads.
__global__ void __launch_bounds__(128)
rvq_update(const float* __restrict__ cbq, const int* __restrict__ idx_ws,
           float* __restrict__ residual, unsigned int* __restrict__ res_hi,
           unsigned int* __restrict__ res_lo, float* __restrict__ qout,
           float* __restrict__ partial, int q) {
  const int row = blockIdx.x;
  const int t   = threadIdx.x;
  const int ci  = idx_ws[row * NQ + q];
  const float4* cr = (const float4*)(cbq + (size_t)ci * D_);
  float4* rr = (float4*)(residual + (size_t)row * D_);
  float4* qr = (float4*)(qout     + (size_t)row * D_);
  unsigned int* hr = res_hi + (size_t)row * DU;
  unsigned int* lr = res_lo + (size_t)row * DU;

  float4 c4 = cr[t];
  float4 r4 = rr[t];
  float4 n4 = make_float4(r4.x - c4.x, r4.y - c4.y, r4.z - c4.z, r4.w - c4.w);
  float4 qo = qr[t];
  qo.x += c4.x; qo.y += c4.y; qo.z += c4.z; qo.w += c4.w;
  rr[t] = n4;
  qr[t] = qo;
  float l0 = n4.x - trunc_bf(n4.x), l1 = n4.y - trunc_bf(n4.y);
  float l2 = n4.z - trunc_bf(n4.z), l3 = n4.w - trunc_bf(n4.w);
  hr[2 * t]     = pack_bf(n4.x, n4.y);
  hr[2 * t + 1] = pack_bf(n4.z, n4.w);
  lr[2 * t]     = pack_bf(l0, l1);
  lr[2 * t + 1] = pack_bf(l2, l3);

  __shared__ float sm[128];
  sm[t] = n4.x * n4.x + n4.y * n4.y + n4.z * n4.z + n4.w * n4.w;
  __syncthreads();
#pragma unroll
  for (int s = 64; s > 0; s >>= 1) {
    if (t < s) sm[t] += sm[t + s];
    __syncthreads();
  }
  if (t == 0) partial[row] = sm[0];
}

// ---------------------------------------------------------------------------
// loss: deterministic single-block reduction of ROWS partials, scaled to mean.
__global__ void __launch_bounds__(256)
rvq_loss(const float* __restrict__ partial, float* __restrict__ loss, int q) {
  __shared__ float sm[256];
  const int t = threadIdx.x;
  float s = 0.f;
  for (int i = t; i < ROWS; i += 256) s += partial[i];
  sm[t] = s;
  __syncthreads();
#pragma unroll
  for (int str = 128; str > 0; str >>= 1) {
    if (t < str) sm[t] += sm[t + str];
    __syncthreads();
  }
  if (t == 0) loss[q] = sm[0] * (1.0f / ((float)ROWS * (float)D_));
}

// ---------------------------------------------------------------------------
extern "C" void kernel_launch(void* const* d_in, const int* in_sizes, int n_in,
                              void* d_out, int out_size, void* d_ws, size_t ws_size,
                              hipStream_t stream) {
  (void)in_sizes; (void)n_in; (void)out_size; (void)ws_size;
  const float* x   = (const float*)d_in[0];   // [8,1024,512]
  const float* cbs = (const float*)d_in[1];   // [8,1024,512]

  float* out_q    = (float*)d_out;                        // [ROWS*D_]
  float* out_idx  = out_q + (size_t)ROWS * D_;            // [ROWS*NQ]
  float* out_loss = out_idx + (size_t)ROWS * NQ;          // [NQ]

  char* w = (char*)d_ws;
  float* residual      = (float*)w;        w += (size_t)ROWS * D_ * 4;        // 16 MB
  unsigned int* res_hi = (unsigned int*)w; w += (size_t)ROWS * DU * 4;        //  8 MB
  unsigned int* res_lo = (unsigned int*)w; w += (size_t)ROWS * DU * 4;        //  8 MB
  unsigned int* cb_hi  = (unsigned int*)w; w += (size_t)NQ * C_ * DU * 4;     //  8 MB
  unsigned int* cb_lo  = (unsigned int*)w; w += (size_t)NQ * C_ * DU * 4;     //  8 MB
  float* e2            = (float*)w;        w += (size_t)NQ * C_ * 4;          // 32 KB
  int* idx_ws          = (int*)w;          w += (size_t)ROWS * NQ * 4;        // 256 KB
  float* partial       = (float*)w;        w += (size_t)ROWS * 4;             // 32 KB

  rvq_init<<<ROWS, 128, 0, stream>>>(x, residual, res_hi, res_lo, out_q);
  rvq_prep<<<NQ * C_, 128, 0, stream>>>(cbs, cb_hi, cb_lo, e2);

  for (int q = 0; q < NQ; ++q) {
    rvq_argmin<<<ROWS / 32, 256, 0, stream>>>(
        res_hi, res_lo,
        cb_hi + (size_t)q * C_ * DU, cb_lo + (size_t)q * C_ * DU,
        e2 + (size_t)q * C_, idx_ws, out_idx, q);
    rvq_update<<<ROWS, 128, 0, stream>>>(
        cbs + (size_t)q * C_ * D_, idx_ws, residual, res_hi, res_lo,
        out_q, partial, q);
    rvq_loss<<<1, 256, 0, stream>>>(partial, out_loss, q);
  }
}
